// temporal_contrastive_loss_64372969832690
// MI455X (gfx1250) — matrix operand order repaired and checked by reading, not compiled, not run on hardware
//
#include <hip/hip_runtime.h>
#include <hip/hip_bf16.h>

typedef __attribute__((ext_vector_type(16))) _Float16     v16h;
typedef __attribute__((ext_vector_type(8)))  float        v8f;
typedef __attribute__((ext_vector_type(8)))  unsigned int v8u;

// Problem constants (reference: img_feats (8, 2048, 256) fp32)
#define BB 8
#define SS 2048
#define DD 256
#define TEMP_INV 10.0f   // 1 / 0.1
#define NCC 4            // column chunks (512 cols each)
#define NANCH (BB * SS)

// ---------------------------------------------------------------------------
// K1: L2-normalize each D=256 row in fp32, store f16. One wave per row.
// ---------------------------------------------------------------------------
__global__ __launch_bounds__(256) void k_normalize(const float* __restrict__ x,
                                                   _Float16* __restrict__ xh) {
    const int lane = threadIdx.x & 31;
    const int wid  = threadIdx.x >> 5;
    const size_t row = (size_t)blockIdx.x * 8 + wid;   // 8 rows per block
    const float* xr = x + row * DD;

    float v[8];
    float ss = 0.0f;
#pragma unroll
    for (int j = 0; j < 8; ++j) {
        v[j] = xr[lane + 32 * j];
        ss += v[j] * v[j];
    }
#pragma unroll
    for (int off = 16; off > 0; off >>= 1)
        ss += __shfl_xor(ss, off, 32);

    const float inv = 1.0f / fmaxf(sqrtf(ss), 1e-12f);
    _Float16* hr = xh + row * DD;
#pragma unroll
    for (int j = 0; j < 8; ++j)
        hr[lane + 32 * j] = (_Float16)(v[j] * inv);
}

// ---------------------------------------------------------------------------
// K2: WMMA row-sum of exp(10 * X Xᵀ).
// Block = (batch b, 64-row block rb, 512-col chunk cc); 4 waves, wave w owns
// rows rb*64 + w*16 .. +15. All 4 waves sweep the SAME column tiles in step,
// so their B-tile loads share WGP$ lines (4x less L2 traffic than 16-row
// blocks). Per column tile: full 8KB B tile loaded into 64 VGPRs up front,
// then 8 chained v_wmma_f32_16x16x32_f16; next tile prefetched
// (global_prefetch_b8) before the chain.
//
// Fragment layouts (CDNA5 ISA 7.12.2, wave32, 16-bit):
//   A (16x32, MxK): lane L -> M = L&15, half = L>>4;
//       VGPR v: K = 32*kc + 16*(v>>2) + 2*(v&3) + 8*half   (pair K,K+1)
//   B (32x16, KxN): lane L -> N = L&15, half = L>>4;
//       VGPR v: K = 32*kc + 16*(v>>2) + 2*(v&3) + 8*half   (pair K,K+1)
//   C (16x16 f32): lane L, VGPR r -> (M = r + 8*(L>>4), N = L&15)
// ---------------------------------------------------------------------------
__global__ __launch_bounds__(128) void k_rowsum(const _Float16* __restrict__ xh,
                                                float* __restrict__ rowsum_part) {
    const int cc = blockIdx.x & (NCC - 1);        // column chunk 0..3
    const int rb = (blockIdx.x >> 2) & 31;        // 64-row block 0..31
    const int b  = blockIdx.x >> 7;               // batch 0..7

    const int lane = threadIdx.x & 31;
    const int wid  = threadIdx.x >> 5;
    const int m    = lane & 15;
    const int half = lane >> 4;

    const int rowBase = rb * 64 + wid * 16;
    const int colBase = cc * (SS / NCC);          // 512-col chunk

    const _Float16* xb = xh + (size_t)b * SS * DD;

    // Preload A fragments for this wave's 16-row tile (8 K-chunks of 32).
    v16h afrag[8];
    {
        const _Float16* arow = xb + (size_t)(rowBase + m) * DD;
#pragma unroll
        for (int kc = 0; kc < 8; ++kc) {
            v8u au;
#pragma unroll
            for (int v = 0; v < 8; ++v) {
                const int K = kc * 32 + ((v >> 2) * 16) + ((v & 3) * 2) + half * 8;
                au[v] = *(const unsigned int*)(arow + K);
            }
            afrag[kc] = __builtin_bit_cast(v16h, au);
        }
    }

    v8f acc = {};   // per-lane partial exp-sums, indexed by C-row r

    for (int ct = 0; ct < (SS / NCC) / 16; ++ct) {
        const int colN = colBase + ct * 16;
        const _Float16* brow = xb + (size_t)(colN + m) * DD;  // N = lane&15

        // Load the full 16x256 B tile into registers (one big load clause).
        v8u btile[8];
#pragma unroll
        for (int kc = 0; kc < 8; ++kc) {
#pragma unroll
            for (int v = 0; v < 8; ++v) {
                const int K = kc * 32 + ((v >> 2) * 16) + ((v & 3) * 2) + half * 8;
                btile[kc][v] = *(const unsigned int*)(brow + K);
            }
        }

        // Prefetch next column tile: 32 lanes cover 16 rows x 2 half-rows
        // (each 128 elements = 256B) = all 32 cache lines of the next tile.
        if (ct + 1 < (SS / NCC) / 16) {
            const _Float16* p =
                xb + (size_t)(colN + 16 + (lane & 15)) * DD + (lane >> 4) * 128;
            __builtin_prefetch((const void*)p, 0, 3);
        }

        v8f c = {};
#pragma unroll
        for (int kc = 0; kc < 8; ++kc) {
            v16h bfrag = __builtin_bit_cast(v16h, btile[kc]);
            c = __builtin_amdgcn_wmma_f32_16x16x32_f16(
                    /*neg_a=*/false, afrag[kc], /*neg_b=*/false, bfrag,
                    /*c_mod=*/(short)0, c, /*reuse_a=*/false, /*reuse_b=*/false);
        }
#pragma unroll
        for (int r = 0; r < 8; ++r)
            acc[r] += __expf(c[r] * TEMP_INV);
    }

    // Reduce each wave's 16 row sums across its 16 lanes (per half).
    __shared__ float smem[4 * 32 * 8];
#pragma unroll
    for (int r = 0; r < 8; ++r)
        smem[(wid * 32 + lane) * 8 + r] = acc[r];
    __syncthreads();

    if (threadIdx.x < 64) {
        const int w  = threadIdx.x >> 4;     // which wave's tile
        const int mm = threadIdx.x & 15;     // row within tile, M = r + 8*half
        const int hs = mm >> 3;
        const int r  = mm & 7;
        float s = 0.0f;
        for (int l = 0; l < 16; ++l)
            s += smem[(w * 32 + hs * 16 + l) * 8 + r];
        const size_t idx = (size_t)b * SS + rb * 64 + w * 16 + mm;
        rowsum_part[(size_t)cc * NANCH + idx] = s;
    }
}

// ---------------------------------------------------------------------------
// K3: per-anchor band fixup + loss. One wave per anchor (b,s):
//   rowsum = fixed-order sum of the 4 column-chunk partials
//   band   = sum_{|s-t|<=9} exp(10*<x_s,x_t>)   (fp32, from the same f16 data)
//   neg    = rowsum - band
//   loss   = sum over positives t=s±1 of log(exp(sim)+neg) - sim
// ---------------------------------------------------------------------------
__global__ __launch_bounds__(256) void k_band(const _Float16* __restrict__ xh,
                                              const float* __restrict__ rowsum_part,
                                              float* __restrict__ loss_part) {
    const int lane = threadIdx.x & 31;
    const int wid  = threadIdx.x >> 5;
    const int a    = blockIdx.x * 8 + wid;   // anchor index in [0, B*S)
    const int b    = a >> 11;                // / S
    const int s    = a & (SS - 1);

    const _Float16* xb = xh + (size_t)b * SS * DD;

    float xs[8];
#pragma unroll
    for (int j = 0; j < 8; ++j)
        xs[j] = (float)xb[(size_t)s * DD + lane + 32 * j];

    int t0 = s - 9; if (t0 < 0) t0 = 0;
    int t1 = s + 9; if (t1 > SS - 1) t1 = SS - 1;

    float band = 0.0f, simL = 0.0f, simR = 0.0f;
    for (int t = t0; t <= t1; ++t) {
        float d = 0.0f;
#pragma unroll
        for (int j = 0; j < 8; ++j)
            d += xs[j] * (float)xb[(size_t)t * DD + lane + 32 * j];
#pragma unroll
        for (int off = 16; off > 0; off >>= 1)
            d += __shfl_xor(d, off, 32);
        const float sim = d * TEMP_INV;
        band += __expf(sim);
        if (t == s - 1) simL = sim;
        if (t == s + 1) simR = sim;
    }

    float rs = rowsum_part[a]
             + rowsum_part[1 * NANCH + a]
             + rowsum_part[2 * NANCH + a]
             + rowsum_part[3 * NANCH + a];

    float neg = rs - band;
    if (neg < 0.0f) neg = 0.0f;

    float lp = 0.0f;
    const bool has_neg = (s >= 10) || (s + 10 <= SS - 1);
    if (has_neg) {
        if (s > 0)      lp += __logf(__expf(simL) + neg) - simL;
        if (s < SS - 1) lp += __logf(__expf(simR) + neg) - simR;
    }
    if (lane == 0) loss_part[a] = lp;
}

// ---------------------------------------------------------------------------
// K4: deterministic final reduction over B*S partials, divide by n_valid.
// ---------------------------------------------------------------------------
__global__ __launch_bounds__(256) void k_finalize(const float* __restrict__ loss_part,
                                                  float* __restrict__ out) {
    __shared__ float sm[256];
    float s = 0.0f;
    for (int i = threadIdx.x; i < NANCH; i += 256)
        s += loss_part[i];
    sm[threadIdx.x] = s;
    __syncthreads();
    for (int st = 128; st > 0; st >>= 1) {
        if ((int)threadIdx.x < st) sm[threadIdx.x] += sm[threadIdx.x + st];
        __syncthreads();
    }
    if (threadIdx.x == 0) {
        const float n_valid = (float)BB * (2.0f * (float)SS - 2.0f); // all anchors valid (S >= 20)
        out[0] = sm[0] / n_valid;
    }
}

// ---------------------------------------------------------------------------
extern "C" void kernel_launch(void* const* d_in, const int* in_sizes, int n_in,
                              void* d_out, int out_size, void* d_ws, size_t ws_size,
                              hipStream_t stream) {
    (void)in_sizes; (void)n_in; (void)out_size; (void)ws_size;

    const float* x   = (const float*)d_in[0];
    float*       out = (float*)d_out;

    char* ws = (char*)d_ws;
    _Float16* xh          = (_Float16*)ws;                              // B*S*D f16 = 8 MB
    float*    rowsum_part = (float*)(ws + (size_t)BB * SS * DD * 2);    // NCC * B*S floats = 256 KB
    float*    loss_part   = rowsum_part + (size_t)NCC * NANCH;          // B*S floats

    k_normalize<<<NANCH / 8, 256, 0, stream>>>(x, xh);
    k_rowsum  <<<BB * 32 * NCC, 128, 0, stream>>>(xh, rowsum_part);
    k_band    <<<NANCH / 8, 256, 0, stream>>>(xh, rowsum_part, loss_part);
    k_finalize<<<1, 256, 0, stream>>>(loss_part, out);
}
